// DVAE_53927609369221
// MI455X (gfx1250) — matrix operand order repaired and checked by reading, not compiled, not run on hardware
//
#include <hip/hip_runtime.h>
#include <hip/hip_bf16.h>
#include <stdint.h>

// ---------------- problem constants ----------------
#define BATCH 2048
#define MAXN  32
#define NVT   10
#define HS    501
#define VS    533          // HS + MAXN
#define NZ    56
#define KP    512          // padded hidden (K) dimension
#define NGH   1536         // padded 3*HS
#define GH_LD 1536

typedef __attribute__((ext_vector_type(16))) _Float16 v16h;
typedef __attribute__((ext_vector_type(8)))  _Float16 v8h;
typedef __attribute__((ext_vector_type(8)))  float    v8f;

__device__ __forceinline__ float sigmoidf_(float x) { return 1.0f / (1.0f + __expf(-x)); }

// ---- WMMA fragment loads (wave32, 16x16x32 f16) ----
// A (16xK tile from row-major MxK, lda in halves):
//   lanes 0-15  hold row m0+lane,       K = kc+{0..7, 16..23}
//   lanes 16-31 hold row m0+(lane-16),  K = kc+{8..15, 24..31}
__device__ __forceinline__ v16h ldfragA(const _Float16* base, int ld, int row0, int kc, int lane) {
    int r = row0 + (lane & 15);
    const _Float16* q = base + (size_t)r * ld + kc + ((lane >> 4) & 1) * 8;
    v8h lo = *(const v8h*)(q);
    v8h hi = *(const v8h*)(q + 16);
    return __builtin_shufflevector(lo, hi, 0,1,2,3,4,5,6,7,8,9,10,11,12,13,14,15);
}

// B (Kx16 tile, sourced from row-major Nx K weight, i.e. B^T):
//   lanes 0-15  hold col n0+lane,       K = kc+{0..7, 16..23}
//   lanes 16-31 hold col n0+(lane-16),  K = kc+{8..15, 24..31}
__device__ __forceinline__ v16h ldfragB(const _Float16* base, int ld, int n0, int kc, int lane) {
    int r = n0 + (lane & 15);
    const _Float16* q = base + (size_t)r * ld + kc + ((lane >> 4) & 1) * 8;
    v8h lo = *(const v8h*)(q);
    v8h hi = *(const v8h*)(q + 16);
    return __builtin_shufflevector(lo, hi, 0,1,2,3,4,5,6,7,8,9,10,11,12,13,14,15);
}

// ---------------- weight pre-conversion ----------------
__global__ void pad_whh_kernel(const float* __restrict__ W_hh, _Float16* __restrict__ Wp) {
    int tid = blockIdx.x * blockDim.x + threadIdx.x;      // NGH * KP threads
    int r = tid >> 9, c = tid & 511;
    float v = (r < 3 * HS && c < HS) ? W_hh[r * HS + c] : 0.0f;
    Wp[tid] = (_Float16)v;
}

__global__ void pad_w501_kernel(const float* __restrict__ W, _Float16* __restrict__ Wp) {
    int tid = blockIdx.x * blockDim.x + threadIdx.x;      // KP * KP threads
    int r = tid >> 9, c = tid & 511;
    float v = (r < HS && c < HS) ? W[r * VS + c] : 0.0f;  // first HS cols of (HS x VS)
    Wp[tid] = (_Float16)v;
}

// ---------------- step 1: predecessor aggregation ----------------
// H[b,j] = sum_{n<v} adj[b,n,v] * G[b,n,j]
__global__ void aggregate_kernel(const uint8_t* __restrict__ adj, const _Float16* __restrict__ G,
                                 float* __restrict__ H, _Float16* __restrict__ Hh, int v) {
    int tid = blockIdx.x * blockDim.x + threadIdx.x;      // BATCH * KP threads
    int b = tid >> 9, j = tid & 511;
    const uint8_t*  arow = adj + (size_t)b * (MAXN * MAXN);
    const _Float16* gb   = G + (size_t)b * MAXN * KP + j;
    float s = 0.0f;
    for (int n = 0; n < v; ++n)
        if (arow[n * MAXN + v]) s += (float)gb[n * KP];
    H[tid]  = s;
    Hh[tid] = (_Float16)s;
}

// ---------------- step 2: gh = H @ W_hh^T (WMMA) ----------------
__global__ void __launch_bounds__(256) gemm_gh_kernel(const _Float16* __restrict__ Hh,
                                                      const _Float16* __restrict__ Whh,
                                                      const float* __restrict__ b_hh,
                                                      float* __restrict__ gh) {
    int lane = threadIdx.x & 31, wave = threadIdx.x >> 5;
    int m0 = blockIdx.x * 32 + (wave >> 2) * 16;          // 64 blocks in M
    int n0 = blockIdx.y * 256 + (wave & 3) * 64;          // 6 blocks in N (1536)
    v8f acc[4] = {};
    for (int kc = 0; kc < KP; kc += 32) {
        v16h a = ldfragA(Hh, KP, m0, kc, lane);
#pragma unroll
        for (int t = 0; t < 4; ++t) {
            v16h bm = ldfragB(Whh, KP, n0 + t * 16, kc, lane);
            acc[t] = __builtin_amdgcn_wmma_f32_16x16x32_f16(false, a, false, bm,
                                                            (short)0, acc[t], false, false);
        }
    }
    int col = lane & 15;
    int rbase = m0 + ((lane >> 4) << 3);
#pragma unroll
    for (int t = 0; t < 4; ++t) {
        int j = n0 + t * 16 + col;
        if (j < 3 * HS) {
            float bias = b_hh[j];
#pragma unroll
            for (int i = 0; i < 8; ++i)
                gh[(size_t)(rbase + i) * GH_LD + j] = acc[t][i] + bias;
        }
    }
}

// ---------------- step 3: GRU elementwise ----------------
__global__ void gru_kernel(const int* __restrict__ node_types, const float* __restrict__ W_ih,
                           const float* __restrict__ b_ih, const float* __restrict__ H,
                           const float* __restrict__ gh, float* __restrict__ Hv,
                           _Float16* __restrict__ Hvh, int v) {
    int tid = blockIdx.x * blockDim.x + threadIdx.x;      // BATCH * KP threads
    int b = tid >> 9, j = tid & 511;
    float val = 0.0f;
    if (j < HS) {
        int t = node_types[b * MAXN + v];
        float gir = W_ih[j * NVT + t] + b_ih[j];
        float giz = W_ih[(HS + j) * NVT + t] + b_ih[HS + j];
        float gin = W_ih[(2 * HS + j) * NVT + t] + b_ih[2 * HS + j];
        const float* ghrow = gh + (size_t)b * GH_LD;
        float ghr = ghrow[j], ghz = ghrow[HS + j], ghn = ghrow[2 * HS + j];
        float h = H[tid];
        float r = sigmoidf_(gir + ghr);
        float z = sigmoidf_(giz + ghz);
        float n = tanhf(gin + r * ghn);
        val = (1.0f - z) * n + z * h;
    }
    Hv[tid]  = val;
    Hvh[tid] = (_Float16)val;
}

// ---------------- step 4: gated row cache (two WMMA GEMMs fused) ----------------
// G[b,v,j] = sigmoid(Hv@WgH^T + bg[j] + Wg[j,HS+v]) * (Hv@WmH^T + Wm[j,HS+v])
__global__ void __launch_bounds__(256) gemm_gate_kernel(const _Float16* __restrict__ Hvh,
                                                        const _Float16* __restrict__ Wgp,
                                                        const _Float16* __restrict__ Wmp,
                                                        const float* __restrict__ Wg,
                                                        const float* __restrict__ bg,
                                                        const float* __restrict__ Wm,
                                                        _Float16* __restrict__ G, int v) {
    int lane = threadIdx.x & 31, wave = threadIdx.x >> 5;
    int m0 = blockIdx.x * 32 + (wave >> 2) * 16;          // 64 blocks in M
    int n0 = blockIdx.y * 256 + (wave & 3) * 64;          // 2 blocks in N (512)
    v8f ag[4] = {}, am[4] = {};
    for (int kc = 0; kc < KP; kc += 32) {
        v16h a = ldfragA(Hvh, KP, m0, kc, lane);
#pragma unroll
        for (int t = 0; t < 4; ++t) {
            v16h bgf = ldfragB(Wgp, KP, n0 + t * 16, kc, lane);
            ag[t] = __builtin_amdgcn_wmma_f32_16x16x32_f16(false, a, false, bgf,
                                                           (short)0, ag[t], false, false);
            v16h bmf = ldfragB(Wmp, KP, n0 + t * 16, kc, lane);
            am[t] = __builtin_amdgcn_wmma_f32_16x16x32_f16(false, a, false, bmf,
                                                           (short)0, am[t], false, false);
        }
    }
    int col = lane & 15;
    int rbase = m0 + ((lane >> 4) << 3);
#pragma unroll
    for (int t = 0; t < 4; ++t) {
        int j = n0 + t * 16 + col;
        if (j < HS) {
            float biasg = bg[j] + Wg[j * VS + HS + v];
            float biasm = Wm[j * VS + HS + v];
#pragma unroll
            for (int i = 0; i < 8; ++i) {
                float gv = sigmoidf_(ag[t][i] + biasg) * (am[t][i] + biasm);
                G[((size_t)(rbase + i) * MAXN + v) * KP + j] = (_Float16)gv;
            }
        } else {
#pragma unroll
            for (int i = 0; i < 8; ++i)
                G[((size_t)(rbase + i) * MAXN + v) * KP + j] = (_Float16)0.0f;
        }
    }
}

// ---------------- final heads: mu / logvar ----------------
__global__ void head_kernel(const float* __restrict__ Hg, const float* __restrict__ W1,
                            const float* __restrict__ b1, const float* __restrict__ W2,
                            const float* __restrict__ b2, float* __restrict__ out) {
    int tid = blockIdx.x * blockDim.x + threadIdx.x;      // BATCH * 2 * NZ threads
    int b = tid / (2 * NZ);
    int r = tid % (2 * NZ);
    int which = r / NZ, i = r % NZ;
    const float* W  = which ? W2 : W1;
    const float* bb = which ? b2 : b1;
    const float* h  = Hg + (size_t)b * KP;
    const float* wr = W + i * HS;
    float s = bb[i];
    for (int j = 0; j < HS; ++j) s += h[j] * wr[j];
    out[(size_t)which * BATCH * NZ + (size_t)b * NZ + i] = s;
}

// ---------------- host launch ----------------
extern "C" void kernel_launch(void* const* d_in, const int* in_sizes, int n_in,
                              void* d_out, int out_size, void* d_ws, size_t ws_size,
                              hipStream_t stream) {
    (void)in_sizes; (void)n_in; (void)out_size; (void)ws_size;
    const int*     node_types = (const int*)d_in[0];
    const uint8_t* adj        = (const uint8_t*)d_in[1];   // jnp bool -> 1 byte
    const float*   W_ih       = (const float*)d_in[2];
    const float*   W_hh       = (const float*)d_in[3];
    const float*   b_ih       = (const float*)d_in[4];
    const float*   b_hh       = (const float*)d_in[5];
    const float*   Wg         = (const float*)d_in[6];
    const float*   bg         = (const float*)d_in[7];
    const float*   Wm         = (const float*)d_in[8];
    const float*   W1         = (const float*)d_in[9];
    const float*   b1         = (const float*)d_in[10];
    const float*   W2         = (const float*)d_in[11];
    const float*   b2         = (const float*)d_in[12];

    char* ws = (char*)d_ws;
    _Float16* Whh_h = (_Float16*)(ws + 0);                 // 1536*512*2  = 1,572,864
    _Float16* Wg_h  = (_Float16*)(ws + 1572864);           //  512*512*2  =   524,288
    _Float16* Wm_h  = (_Float16*)(ws + 2097152);           //  512*512*2  =   524,288
    float*    Hbuf  = (float*)   (ws + 2621440);           // 2048*512*4  = 4,194,304
    _Float16* Hh    = (_Float16*)(ws + 6815744);           // 2048*512*2  = 2,097,152
    float*    ghb   = (float*)   (ws + 8912896);           // 2048*1536*4 = 12,582,912
    float*    Hv    = (float*)   (ws + 21495808);          // 2048*512*4  = 4,194,304
    _Float16* Hvh   = (_Float16*)(ws + 25690112);          // 2048*512*2  = 2,097,152
    _Float16* G     = (_Float16*)(ws + 27787264);          // 2048*32*512*2 = 67,108,864
    // total workspace use: ~94.9 MB

    // one-time (per launch) weight conversion to zero-padded f16
    pad_whh_kernel<<<(NGH * KP) / 256, 256, 0, stream>>>(W_hh, Whh_h);
    pad_w501_kernel<<<(KP * KP) / 256, 256, 0, stream>>>(Wg, Wg_h);
    pad_w501_kernel<<<(KP * KP) / 256, 256, 0, stream>>>(Wm, Wm_h);

    for (int v = 0; v < MAXN; ++v) {
        aggregate_kernel<<<(BATCH * KP) / 256, 256, 0, stream>>>(adj, G, Hbuf, Hh, v);
        gemm_gh_kernel<<<dim3(BATCH / 32, NGH / 256), 256, 0, stream>>>(Hh, Whh_h, b_hh, ghb);
        gru_kernel<<<(BATCH * KP) / 256, 256, 0, stream>>>(node_types, W_ih, b_ih,
                                                           Hbuf, ghb, Hv, Hvh, v);
        if (v < MAXN - 1)  // gated row of the last node is never consumed
            gemm_gate_kernel<<<dim3(BATCH / 32, KP / 256), 256, 0, stream>>>(
                Hvh, Wg_h, Wm_h, Wg, bg, Wm, G, v);
    }
    // Hg = H_all[:, 31] = Hv after the final step
    head_kernel<<<(BATCH * 2 * NZ) / 256, 256, 0, stream>>>(Hv, W1, b1, W2, b2, (float*)d_out);
}
